// CurriculumLearningGuidedDynamicLoss_44092134261357
// MI455X (gfx1250) — compile-verified, matrix operand order checked by reading, not run on hardware
//
#include <hip/hip_runtime.h>
#include <hip/hip_bf16.h>
#include <math.h>

typedef __attribute__((ext_vector_type(16))) _Float16 v16h;
typedef __attribute__((ext_vector_type(8)))  _Float16 v8h;
typedef __attribute__((ext_vector_type(8)))  float    v8f;

#define NBINS   140
#define NPAD    320     // padded N for the GEMM (sin 0..139, cos 140..279, zeros 280..319)
#define M_TILE  128
#define N_TILE  64
#define K_TILE  32
#define LDH     48      // LDS row stride in halves (96 B, 16B-aligned chunks)

// ---------------------------------------------------------------------------
// 1) Trig table: Bt[r][j] = hann[j] * sin(w_r*j) (r<140) / cos (140<=r<280) / 0
// ---------------------------------------------------------------------------
__global__ __launch_bounds__(256)
void cldl_table_kernel(_Float16* __restrict__ Bt, int n) {
  const int r = blockIdx.x;                       // 0..NPAD-1
  const bool valid = r < 2 * NBINS;
  const int bin = (r < NBINS) ? r : (r - NBINS);
  const float two_pi = 6.28318530717958647692f;
  const float omega = two_pi * (float)(40 + bin) / 1800.0f;  // 2*pi*bpm/(60*FS), FS=30
  const float hann_k = two_pi / (float)(n - 1);
  _Float16* dst = Bt + (size_t)r * n;
  for (int j = threadIdx.x; j < n; j += 256) {
    float val = 0.0f;
    if (valid) {
      float hann = 0.5f - 0.5f * cosf(hann_k * (float)j);
      float ang  = omega * (float)j;
      float tr   = (r < NBINS) ? sinf(ang) : cosf(ang);
      val = hann * tr;
    }
    dst[j] = (_Float16)val;
  }
}

// ---------------------------------------------------------------------------
// 2) Neg-Pearson per row (Pearson is invariant to the reference's z-scoring)
// ---------------------------------------------------------------------------
__global__ __launch_bounds__(256)
void cldl_pearson_kernel(const float* __restrict__ P, const float* __restrict__ T,
                         float* __restrict__ perRow, int n) {
  const int b    = blockIdx.x;
  const int tid  = threadIdx.x;
  const int lane = tid & 31;
  const int wave = tid >> 5;
  const float4* p = (const float4*)(P + (size_t)b * n);
  const float4* t = (const float4*)(T + (size_t)b * n);
  float sx = 0.f, sy = 0.f, sxy = 0.f, sx2 = 0.f, sy2 = 0.f;
  for (int i = tid; i < n / 4; i += 256) {
    float4 x = p[i], y = t[i];
    sx  += x.x + x.y + x.z + x.w;
    sy  += y.x + y.y + y.z + y.w;
    sxy += x.x * y.x + x.y * y.y + x.z * y.z + x.w * y.w;
    sx2 += x.x * x.x + x.y * x.y + x.z * x.z + x.w * x.w;
    sy2 += y.x * y.x + y.y * y.y + y.z * y.z + y.w * y.w;
  }
  #pragma unroll
  for (int m = 16; m; m >>= 1) {
    sx  += __shfl_xor(sx,  m, 32);
    sy  += __shfl_xor(sy,  m, 32);
    sxy += __shfl_xor(sxy, m, 32);
    sx2 += __shfl_xor(sx2, m, 32);
    sy2 += __shfl_xor(sy2, m, 32);
  }
  __shared__ float sm[5][8];
  if (lane == 0) { sm[0][wave] = sx; sm[1][wave] = sy; sm[2][wave] = sxy;
                   sm[3][wave] = sx2; sm[4][wave] = sy2; }
  __syncthreads();
  if (tid == 0) {
    float a = 0, bb = 0, c = 0, d = 0, e = 0;
    #pragma unroll
    for (int w = 0; w < 8; ++w) { a += sm[0][w]; bb += sm[1][w]; c += sm[2][w];
                                  d += sm[3][w]; e += sm[4][w]; }
    float fn  = (float)n;
    float num = fn * c - a * bb;
    float den = sqrtf((fn * d - a * a) * (fn * e - bb * bb));
    perRow[b] = 1.0f - num / den;
  }
}

// ---------------------------------------------------------------------------
// 3) WMMA GEMM (double-buffered, software-pipelined):
//    C[1024][320] = A[1024][8192](f32->f16) x Bt[320][8192]^T
// ---------------------------------------------------------------------------
__global__ __launch_bounds__(256)
void cldl_gemm_kernel(const float* __restrict__ A, const _Float16* __restrict__ Bt,
                      float* __restrict__ C, int K) {
  __shared__ _Float16 lds_a[2][M_TILE][LDH];
  __shared__ _Float16 lds_b[2][N_TILE][LDH];

  const int tid  = threadIdx.x;
  const int lane = tid & 31;
  const int wave = tid >> 5;         // 0..7
  const int wm   = wave & 3;         // wave row   (M dir, 4)
  const int wn   = wave >> 2;        // wave col   (N dir, 2)
  const int m_blk = blockIdx.x * M_TILE;
  const int n_blk = blockIdx.y * N_TILE;

  v8f acc[2][2] = {};

  const int koff  = (lane >> 4) << 3;           // 0 or 8 (A/B K sub-offset)
  const int mrow  = wm * 32 + (lane & 15);
  const int nrow  = wn * 32 + (lane & 15);

  // staging indices (uniform per thread across iterations)
  const int arow0 = tid >> 2;                   // chunk 0 row (0..63)
  const int arow1 = (tid + 256) >> 2;           // chunk 1 row (64..127)
  const int acol  = (tid & 3) << 3;             // 0,8,16,24
  const int brow  = tid >> 2;                   // 0..63
  const int bcol  = (tid & 3) << 3;

  const float*    pa0 = A  + (size_t)(m_blk + arow0) * K + acol;
  const float*    pa1 = A  + (size_t)(m_blk + arow1) * K + acol;
  const _Float16* pb  = Bt + (size_t)(n_blk + brow)  * K + bcol;

  float4 ra[2][2];   // two A chunks x two float4
  v8h    rb;

  // ---- prologue: stage k0 = 0 into buffer 0
  {
    const float4* q0 = (const float4*)pa0;
    const float4* q1 = (const float4*)pa1;
    ra[0][0] = q0[0]; ra[0][1] = q0[1];
    ra[1][0] = q1[0]; ra[1][1] = q1[1];
    rb = *(const v8h*)pb;
    v8h h0, h1;
    h0[0]=(_Float16)ra[0][0].x; h0[1]=(_Float16)ra[0][0].y; h0[2]=(_Float16)ra[0][0].z; h0[3]=(_Float16)ra[0][0].w;
    h0[4]=(_Float16)ra[0][1].x; h0[5]=(_Float16)ra[0][1].y; h0[6]=(_Float16)ra[0][1].z; h0[7]=(_Float16)ra[0][1].w;
    h1[0]=(_Float16)ra[1][0].x; h1[1]=(_Float16)ra[1][0].y; h1[2]=(_Float16)ra[1][0].z; h1[3]=(_Float16)ra[1][0].w;
    h1[4]=(_Float16)ra[1][1].x; h1[5]=(_Float16)ra[1][1].y; h1[6]=(_Float16)ra[1][1].z; h1[7]=(_Float16)ra[1][1].w;
    *(v8h*)&lds_a[0][arow0][acol] = h0;
    *(v8h*)&lds_a[0][arow1][acol] = h1;
    *(v8h*)&lds_b[0][brow][bcol]  = rb;
  }
  __syncthreads();

  int buf = 0;
  for (int k0 = 0; k0 < K; k0 += K_TILE) {
    const int  kn       = k0 + K_TILE;
    const bool has_next = kn < K;

    // ---- issue next tile's global loads first (latency hides under WMMA)
    if (has_next) {
      const float4* q0 = (const float4*)(pa0 + kn);
      const float4* q1 = (const float4*)(pa1 + kn);
      ra[0][0] = q0[0]; ra[0][1] = q0[1];
      ra[1][0] = q1[0]; ra[1][1] = q1[1];
      rb = *(const v8h*)(pb + kn);
      // prefetch A two tiles ahead into cache
      if (kn + K_TILE < K) __builtin_prefetch(pa0 + kn + K_TILE, 0, 3);
    }

    // ---- fragments from current buffer (ISA 16-bit A layout:
    //      halves 0..7 -> K+0..7, 8..15 -> K+16..23, lanes>=16 add +8)
    v16h af[2], bf[2];
    #pragma unroll
    for (int ti = 0; ti < 2; ++ti) {
      const _Float16* qa = &lds_a[buf][mrow + ti * 16][koff];
      v8h lo = *(const v8h*)(qa);
      v8h hi = *(const v8h*)(qa + 16);
      #pragma unroll
      for (int q = 0; q < 8; ++q) { af[ti][q] = lo[q]; af[ti][q + 8] = hi[q]; }
    }
    #pragma unroll
    for (int tj = 0; tj < 2; ++tj) {
      const _Float16* qb = &lds_b[buf][nrow + tj * 16][koff];
      v8h lo = *(const v8h*)(qb);
      v8h hi = *(const v8h*)(qb + 16);
      #pragma unroll
      for (int q = 0; q < 8; ++q) { bf[tj][q] = lo[q]; bf[tj][q + 8] = hi[q]; }
    }

    #pragma unroll
    for (int ti = 0; ti < 2; ++ti)
      #pragma unroll
      for (int tj = 0; tj < 2; ++tj)
        acc[ti][tj] = __builtin_amdgcn_wmma_f32_16x16x32_f16(
            false, af[ti], false, bf[tj], (short)0, acc[ti][tj], false, false);

    // ---- convert + store next tile into the other buffer
    if (has_next) {
      v8h h0, h1;
      h0[0]=(_Float16)ra[0][0].x; h0[1]=(_Float16)ra[0][0].y; h0[2]=(_Float16)ra[0][0].z; h0[3]=(_Float16)ra[0][0].w;
      h0[4]=(_Float16)ra[0][1].x; h0[5]=(_Float16)ra[0][1].y; h0[6]=(_Float16)ra[0][1].z; h0[7]=(_Float16)ra[0][1].w;
      h1[0]=(_Float16)ra[1][0].x; h1[1]=(_Float16)ra[1][0].y; h1[2]=(_Float16)ra[1][0].z; h1[3]=(_Float16)ra[1][0].w;
      h1[4]=(_Float16)ra[1][1].x; h1[5]=(_Float16)ra[1][1].y; h1[6]=(_Float16)ra[1][1].z; h1[7]=(_Float16)ra[1][1].w;
      *(v8h*)&lds_a[buf ^ 1][arow0][acol] = h0;
      *(v8h*)&lds_a[buf ^ 1][arow1][acol] = h1;
      *(v8h*)&lds_b[buf ^ 1][brow][bcol]  = rb;
    }
    __syncthreads();
    buf ^= 1;
  }

  // --- store C (D layout: VGPR v -> M=v (+8 for lanes>=16), lane&15 -> N)
  const int rbase = m_blk + wm * 32 + ((lane >> 4) << 3);
  const int cbase = n_blk + wn * 32 + (lane & 15);
  #pragma unroll
  for (int ti = 0; ti < 2; ++ti)
    #pragma unroll
    for (int tj = 0; tj < 2; ++tj)
      #pragma unroll
      for (int v = 0; v < 8; ++v)
        C[(size_t)(rbase + ti * 16 + v) * NPAD + (cbase + tj * 16)] = acc[ti][tj][v];
}

// ---------------------------------------------------------------------------
// 4) Per-row spectrum -> L1 norm -> log_softmax -> CE + KL  (1 wave per row)
// ---------------------------------------------------------------------------
__global__ __launch_bounds__(256)
void cldl_post_kernel(const float* __restrict__ C, const int* __restrict__ hr,
                      float* __restrict__ ceRow, float* __restrict__ klRow) {
  const int lane = threadIdx.x & 31;
  const int b    = blockIdx.x * 8 + (threadIdx.x >> 5);
  const float* row = C + (size_t)b * NPAD;

  float ca[5];
  float sum = 0.f;
  #pragma unroll
  for (int i = 0; i < 5; ++i) {
    int k = lane + 32 * i;
    if (k < NBINS) {
      float s = row[k], c = row[NBINS + k];
      ca[i] = s * s + c * c;
    } else ca[i] = 0.f;
    sum += ca[i];
  }
  #pragma unroll
  for (int m = 16; m; m >>= 1) sum += __shfl_xor(sum, m, 32);
  float inv = 1.0f / sum;

  float mx = -1e30f;
  #pragma unroll
  for (int i = 0; i < 5; ++i) {
    int k = lane + 32 * i;
    if (k < NBINS) mx = fmaxf(mx, ca[i] * inv);
  }
  #pragma unroll
  for (int m = 16; m; m >>= 1) mx = fmaxf(mx, __shfl_xor(mx, m, 32));

  float es = 0.f;
  #pragma unroll
  for (int i = 0; i < 5; ++i) {
    int k = lane + 32 * i;
    if (k < NBINS) es += expf(ca[i] * inv - mx);
  }
  #pragma unroll
  for (int m = 16; m; m >>= 1) es += __shfl_xor(es, m, 32);
  const float lse = mx + logf(es);

  const int h = hr[b];
  float ce = 0.f, kl = 0.f;
  const float inv_s2pi = 0.3989422804014327f;   // 1/sqrt(2*pi)
  #pragma unroll
  for (int i = 0; i < 5; ++i) {
    int k = lane + 32 * i;
    if (k < NBINS) {
      float logp = ca[i] * inv - lse;
      if (k == h) ce = -logp;
      float d = (float)k - (float)h;
      float t = expf(-0.5f * d * d) * inv_s2pi;
      t = fmaxf(t, 1e-15f);
      kl += expf(t) * (t - logp);
    }
  }
  #pragma unroll
  for (int m = 16; m; m >>= 1) { ce += __shfl_xor(ce, m, 32); kl += __shfl_xor(kl, m, 32); }
  if (lane == 0) { ceRow[b] = ce; klRow[b] = kl * (1.0f / (float)NBINS); }
}

// ---------------------------------------------------------------------------
// 5) Deterministic final reduce + alpha/beta combine
// ---------------------------------------------------------------------------
__global__ __launch_bounds__(256)
void cldl_final_kernel(const float* __restrict__ pr, const float* __restrict__ ce,
                       const float* __restrict__ kl, const int* __restrict__ epoch_p,
                       float* __restrict__ out, int B) {
  __shared__ float s0[256], s1[256], s2[256];
  const int t = threadIdx.x;
  float a = 0.f, b = 0.f, c = 0.f;
  for (int i = t; i < B; i += 256) { a += pr[i]; b += ce[i]; c += kl[i]; }
  s0[t] = a; s1[t] = b; s2[t] = c;
  __syncthreads();
  for (int st = 128; st; st >>= 1) {
    if (t < st) { s0[t] += s0[t + st]; s1[t] += s1[t + st]; s2[t] += s2[t + st]; }
    __syncthreads();
  }
  if (t == 0) {
    float invB  = 1.0f / (float)B;
    float tloss = s0[0] * invB;
    float cem   = s1[0] * invB;
    float klm   = s2[0] * invB;
    int epoch = *epoch_p;
    float alpha, beta;
    if (epoch > 25) { alpha = 0.05f; beta = 2.0f; }
    else {
      float e = (float)epoch / 25.0f;
      alpha = 0.1f * powf(0.5f, e);
      beta  = powf(2.0f, e);
    }
    out[0] = alpha * tloss + beta * (cem + klm);
  }
}

// ---------------------------------------------------------------------------
extern "C" void kernel_launch(void* const* d_in, const int* in_sizes, int n_in,
                              void* d_out, int out_size, void* d_ws, size_t ws_size,
                              hipStream_t stream) {
  const int*   epoch = (const int*)d_in[0];
  const float* pred  = (const float*)d_in[1];
  const float* targ  = (const float*)d_in[2];
  const int*   hr    = (const int*)d_in[3];
  float*       out   = (float*)d_out;

  const int B = in_sizes[3];              // 1024
  const int N = in_sizes[1] / B;          // 8192

  // workspace layout
  char* ws = (char*)d_ws;
  float*    Cbuf  = (float*)ws;                                  // B x NPAD f32
  size_t    off   = (size_t)B * NPAD * sizeof(float);
  _Float16* Bt    = (_Float16*)(ws + off);                       // NPAD x N f16
  off += (size_t)NPAD * N * sizeof(_Float16);
  float* perRow   = (float*)(ws + off); off += (size_t)B * sizeof(float);
  float* ceRow    = (float*)(ws + off); off += (size_t)B * sizeof(float);
  float* klRow    = (float*)(ws + off);

  cldl_table_kernel<<<NPAD, 256, 0, stream>>>(Bt, N);
  cldl_pearson_kernel<<<B, 256, 0, stream>>>(pred, targ, perRow, N);
  dim3 gg(B / M_TILE, NPAD / N_TILE);
  cldl_gemm_kernel<<<gg, 256, 0, stream>>>(pred, Bt, Cbuf, N);
  cldl_post_kernel<<<B / 8, 256, 0, stream>>>(Cbuf, hr, ceRow, klRow);
  cldl_final_kernel<<<1, 256, 0, stream>>>(perRow, ceRow, klRow, epoch, out, B);
}
